// Self_Attention_49598282334517
// MI455X (gfx1250) — matrix operand order repaired and checked by reading, not compiled, not run on hardware
//
#include <hip/hip_runtime.h>
#include <hip/hip_bf16.h>
#include <stdint.h>

// ---------------- CDNA5 WMMA plumbing (wave32, gfx1250) ----------------
typedef __bf16 v16bf __attribute__((ext_vector_type(16)));
typedef float  v8f   __attribute__((ext_vector_type(8)));
typedef unsigned v4u __attribute__((ext_vector_type(4)));
typedef int      v8i __attribute__((ext_vector_type(8)));
typedef int      v4i __attribute__((ext_vector_type(4)));

struct Frag32B { uint4 lo, hi; };

static __device__ __forceinline__ v16bf frag_cast(uint4 lo, uint4 hi) {
    Frag32B t; t.lo = lo; t.hi = hi;
    return __builtin_bit_cast(v16bf, t);
}

// A-matrix 16x32 bf16 fragment (ISA 7.12.2): lane L holds row (L&15);
// K halves: [kb, kb+8) and [kb+16, kb+24) with kb = (L>=16)*8.
static __device__ __forceinline__ v16bf load_a_frag(const unsigned short* row,
                                                    int kofs, int lane) {
    int kb = (lane >> 4) * 8;
    uint4 lo = *(const uint4*)(row + kofs + kb);
    uint4 hi = *(const uint4*)(row + kofs + kb + 16);
    return frag_cast(lo, hi);
}

// B-matrix 32x16 bf16 fragment: lane L holds column (L&15);
// K halves: [kb, kb+8) and [kb+16, kb+24) with kb = (L>=16)*8.
static __device__ __forceinline__ v16bf load_b_frag(const unsigned short* col,
                                                    int kofs, int lane) {
    int kb = (lane >> 4) * 8;
    uint4 lo = *(const uint4*)(col + kofs + kb);
    uint4 hi = *(const uint4*)(col + kofs + kb + 16);
    return frag_cast(lo, hi);
}

static __device__ __forceinline__ v8f wmma_bf16(v16bf a, v16bf b, v8f c) {
    return __builtin_amdgcn_wmma_f32_16x16x32_bf16(false, a, false, b,
                                                   (short)0, c, false, false);
}

static __device__ __forceinline__ unsigned short f32_to_bf16(float f) {
    unsigned u = __builtin_bit_cast(unsigned, f);
    u += 0x7FFFu + ((u >> 16) & 1u);               // round to nearest even
    return (unsigned short)(u >> 16);
}

// ---------------- Tensor Data Mover (TDM) ----------------
// Device-pass-only check: a successful compile proves the gfx1250 binary
// contains tensor_load_to_lds. Host pass sees a parse-only stub.
#if defined(__HIP_DEVICE_COMPILE__)
#if __has_builtin(__builtin_amdgcn_tensor_load_to_lds)
#define HAVE_TDM 1
#else
#define HAVE_TDM 0
#error "device pass: tensor_load_to_lds builtin not available"
#endif
#else
#define HAVE_TDM 0   // host parse-only
#endif

// D# (ISA 8.3/8.4) for a 2D tile of 4-byte elements; tile_d1 == 0 => 1D tile.
// stride_d0 in 4B units.
static __device__ __forceinline__ void tdm_load_to_lds(unsigned lds_addr,
                                                       const void* gptr,
                                                       unsigned tile_d0,
                                                       unsigned tile_d1,
                                                       unsigned long long stride_d0) {
#if HAVE_TDM
    unsigned long long ga = (unsigned long long)(uintptr_t)gptr;
    v4u g0;
    g0[0] = 1u;                                       // count=1 (valid), user mode
    g0[1] = lds_addr;                                 // lds_addr [63:32]
    g0[2] = (unsigned)ga;                             // global_addr[31:0]
    g0[3] = (unsigned)((ga >> 32) & 0x1FFFFFFu)       // global_addr[56:32]
          | (2u << 30);                               // type=2 ("image")
    unsigned td0 = tile_d0;                           // tensor_dim0 == tile_dim0
    unsigned td1 = tile_d1 ? tile_d1 : 1u;            // tensor_dim1
    v8i g1;
    g1[0] = (int)(2u << 16);                          // data_size=2 (4B), mask=0
    g1[1] = (int)((td0 & 0xFFFFu) << 16);             // tensor_dim0[15:0] @63:48
    g1[2] = (int)((td0 >> 16) | ((td1 & 0xFFFFu) << 16)); // dim0 hi / dim1 lo
    g1[3] = (int)((td1 >> 16) | (tile_d0 << 16));     // dim1 hi / tile_dim0
    g1[4] = (int)tile_d1;                             // tile_dim1 (0 => 1D)
    g1[5] = (int)(stride_d0 & 0xFFFFFFFFu);           // tensor_dim0_stride lo
    g1[6] = (int)((stride_d0 >> 32) & 0xFFFFu);       // stride hi, dim1_stride=0
    g1[7] = 0;
    v4i g2 = {0, 0, 0, 0};
    v4i g3 = {0, 0, 0, 0};
#if __clang_major__ >= 23
    v8i g4 = {0, 0, 0, 0, 0, 0, 0, 0};
    __builtin_amdgcn_tensor_load_to_lds(g0, g1, g2, g3, g4, 0);
#else
    __builtin_amdgcn_tensor_load_to_lds(g0, g1, g2, g3, 0);
#endif
#else
    (void)lds_addr; (void)gptr; (void)tile_d0; (void)tile_d1; (void)stride_d0;
#endif
}

static __device__ __forceinline__ void tdm_wait_le2() {
#if HAVE_TDM
#if __has_builtin(__builtin_amdgcn_s_wait_tensorcnt)
    __builtin_amdgcn_s_wait_tensorcnt(2);
#else
    asm volatile("s_wait_tensorcnt 0x2" ::: "memory");
#endif
#endif
}
static __device__ __forceinline__ void tdm_wait_le0() {
#if HAVE_TDM
#if __has_builtin(__builtin_amdgcn_s_wait_tensorcnt)
    __builtin_amdgcn_s_wait_tensorcnt(0);
#else
    asm volatile("s_wait_tensorcnt 0x0" ::: "memory");
#endif
#endif
}

// ---------------- Problem constants ----------------
#define BSZ 2
#define SSZ 2048
#define DSZ 1024
#define HN  16
#define DH  64
#define NE  1024   // HN*DH

// ---------------- Stage 0: convert to bf16 ----------------
__global__ void cvt_x_kernel(const float* __restrict__ x,
                             unsigned short* __restrict__ xb, int n) {
    int i = blockIdx.x * blockDim.x + threadIdx.x;
    if (i < n) xb[i] = f32_to_bf16(x[i]);
}

// W [3][D][N] f32 -> wT [3][N][D] bf16 (K-major for B-fragments)
__global__ void cvt_w_kernel(const float* __restrict__ w,
                             unsigned short* __restrict__ wT) {
    int i = blockIdx.x * blockDim.x + threadIdx.x;   // < 3*1024*1024
    int m = i >> 20;
    int rem = i & 0xFFFFF;
    int n = rem >> 10;
    int d = rem & 1023;
    wT[i] = f32_to_bf16(w[(m << 20) + (d << 10) + n]);
}

// ---------------- Stage 1: QKV projection GEMM (WMMA bf16) ----------------
// grid (32, 16, 3), block 256 (8 waves). Software-pipelined K loop, unrolled
// x2 with ping-pong fragment sets so loads write their final registers
// directly (no rotation copies, fewer WMMA->VALU hazard NOPs).
__global__ __launch_bounds__(256)
void proj_kernel(const unsigned short* __restrict__ xb,
                 const unsigned short* __restrict__ wT,
                 unsigned short* __restrict__ Qb,
                 unsigned short* __restrict__ Kb,
                 unsigned short* __restrict__ Vt) {
    const int lane = threadIdx.x & 31;
    const int wave = threadIdx.x >> 5;
    const int m    = blockIdx.z;
    const int h    = blockIdx.y;           // 64-col block == one head

    const int arow = blockIdx.x * 128 + wave * 16 + (lane & 15);
    const unsigned short* aptr  = xb + (size_t)arow * DSZ;
    const unsigned short* wcol[4];
    #pragma unroll
    for (int ct = 0; ct < 4; ++ct)
        wcol[ct] = wT + ((size_t)m * NE + h * 64 + ct * 16 + (lane & 15)) * DSZ;

    v8f c[4] = {};

    // Pipeline prologue: chunk 0 -> set A.
    v16bf aA = load_a_frag(aptr, 0, lane);
    v16bf bA[4];
    #pragma unroll
    for (int ct = 0; ct < 4; ++ct) bA[ct] = load_b_frag(wcol[ct], 0, lane);

    // Steady state: chunks kk (set A) and kk+1 (set B); prefetch kk+2 -> A.
    for (int kk = 0; kk < DSZ / 32 - 2; kk += 2) {
        v16bf aB = load_a_frag(aptr, (kk + 1) * 32, lane);
        v16bf bB[4];
        #pragma unroll
        for (int ct = 0; ct < 4; ++ct)
            bB[ct] = load_b_frag(wcol[ct], (kk + 1) * 32, lane);
        #pragma unroll
        for (int ct = 0; ct < 4; ++ct)
            c[ct] = wmma_bf16(aA, bA[ct], c[ct]);

        aA = load_a_frag(aptr, (kk + 2) * 32, lane);
        #pragma unroll
        for (int ct = 0; ct < 4; ++ct)
            bA[ct] = load_b_frag(wcol[ct], (kk + 2) * 32, lane);
        #pragma unroll
        for (int ct = 0; ct < 4; ++ct)
            c[ct] = wmma_bf16(aB, bB[ct], c[ct]);
    }
    // Epilogue: chunks 30 (set A) and 31 (set B).
    {
        v16bf aB = load_a_frag(aptr, (DSZ / 32 - 1) * 32, lane);
        v16bf bB[4];
        #pragma unroll
        for (int ct = 0; ct < 4; ++ct)
            bB[ct] = load_b_frag(wcol[ct], (DSZ / 32 - 1) * 32, lane);
        #pragma unroll
        for (int ct = 0; ct < 4; ++ct)
            c[ct] = wmma_bf16(aA, bA[ct], c[ct]);
        #pragma unroll
        for (int ct = 0; ct < 4; ++ct)
            c[ct] = wmma_bf16(aB, bB[ct], c[ct]);
    }

    const float qscale = (m == 0) ? 0.125f : 1.0f;  // 1/sqrt(64)
    if (m == 2) {
        // V^T layout: row index r maps to consecutive s -> pack pairs into b32.
        #pragma unroll
        for (int ct = 0; ct < 4; ++ct) {
            #pragma unroll
            for (int r = 0; r < 8; r += 2) {
                int row  = blockIdx.x * 128 + wave * 16 + r + 8 * (lane >> 4);
                int bidx = row >> 11;
                int s    = row & (SSZ - 1);
                int dd   = ct * 16 + (lane & 15);
                unsigned lo = f32_to_bf16(c[ct][r]);
                unsigned hi = f32_to_bf16(c[ct][r + 1]);
                size_t bh = (size_t)(bidx * HN + h);
                *(unsigned*)&Vt[(bh * DH + dd) * SSZ + s] = lo | (hi << 16);
            }
        }
    } else {
        #pragma unroll
        for (int ct = 0; ct < 4; ++ct) {
            #pragma unroll
            for (int r = 0; r < 8; ++r) {
                int row  = blockIdx.x * 128 + wave * 16 + r + 8 * (lane >> 4);
                int bidx = row >> 11;
                int s    = row & (SSZ - 1);
                int dd   = ct * 16 + (lane & 15);
                unsigned short bf = f32_to_bf16(c[ct][r] * qscale);
                size_t bh = (size_t)(bidx * HN + h);
                if (m == 0) Qb[(bh * SSZ + s) * DH + dd] = bf;
                else        Kb[(bh * SSZ + s) * DH + dd] = bf;
            }
        }
    }
}

// ---------------- Stage 2: flash attention (WMMA bf16, f32 softmax) --------
// grid (16, 16, 2) = (qblock, h, b), block 256 (8 waves x 16 query rows).
// K/V blocks staged by the Tensor Data Mover, double-buffered so the TDM
// fetch of block kb+1 overlaps the compute of block kb (TENSORcnt <= 2).
__global__ __launch_bounds__(256)
void attn_kernel(const unsigned short* __restrict__ Qb,
                 const unsigned short* __restrict__ Kb,
                 const unsigned short* __restrict__ Vt,
                 float* __restrict__ out) {
    __shared__ __align__(16) unsigned short sK[2][64 * 64];   // [buf][key][d]
    __shared__ __align__(16) unsigned short sV[2][64 * 64];   // [buf][d][key]
    __shared__ __align__(16) unsigned short sP[8 * 16 * 64];  // per-wave P

    const int tid = threadIdx.x, lane = tid & 31, wave = tid >> 5;
    const int qblk = blockIdx.x, h = blockIdx.y, b = blockIdx.z;
    const int bh = b * HN + h;

    // Q A-fragments, held for the whole kernel (Q is pre-scaled by 1/8).
    const int qrow = qblk * 128 + wave * 16 + (lane & 15);
    const unsigned short* qptr = Qb + ((size_t)bh * SSZ + qrow) * DH;
    v16bf aq[2];
    #pragma unroll
    for (int dc = 0; dc < 2; ++dc) aq[dc] = load_a_frag(qptr, dc * 32, lane);

    v8f o[4] = {};
    float mrow[8], lrow[8];
    #pragma unroll
    for (int r = 0; r < 8; ++r) { mrow[r] = -1e30f; lrow[r] = 0.0f; }

    const unsigned short* Kblk0 = Kb + (size_t)bh * SSZ * DH;
    const unsigned short* Vbase = Vt + (size_t)bh * DH * SSZ;
    unsigned short* myP = sP + wave * 16 * 64;

    // Prologue: TDM loads for block 0.
    if (wave == 0) {
        tdm_load_to_lds((unsigned)(uintptr_t)(void*)sK[0], Kblk0,
                        2048u, 0u, 2048ull);
        tdm_load_to_lds((unsigned)(uintptr_t)(void*)sV[0], Vbase,
                        32u, 64u, 1024ull);
    }

    for (int kb = 0; kb < SSZ / 64; ++kb) {
        const int cur = kb & 1;
        if (wave == 0) {
            if (kb + 1 < SSZ / 64) {
                tdm_load_to_lds((unsigned)(uintptr_t)(void*)sK[1 - cur],
                                Kblk0 + (size_t)(kb + 1) * 64 * DH,
                                2048u, 0u, 2048ull);
                tdm_load_to_lds((unsigned)(uintptr_t)(void*)sV[1 - cur],
                                Vbase + (size_t)(kb + 1) * 64,
                                32u, 64u, 1024ull);
                tdm_wait_le2();   // block kb complete; kb+1 still in flight
            } else {
                tdm_wait_le0();
            }
        }
        __syncthreads();          // block kb visible to all waves
        const unsigned short* cK = sK[cur];
        const unsigned short* cV = sV[cur];

        // S = Q K^T (scaled via Q)
        v8f st[4];
        #pragma unroll
        for (int kt = 0; kt < 4; ++kt) {
            v8f acc = {};
            #pragma unroll
            for (int dc = 0; dc < 2; ++dc) {
                const unsigned short* kcol = cK + (kt * 16 + (lane & 15)) * 64;
                v16bf bk = load_b_frag(kcol, dc * 32, lane);
                acc = wmma_bf16(aq[dc], bk, acc);
            }
            st[kt] = acc;
        }

        // Online softmax: row reductions across the 16-lane half (wave32-safe).
        float al[8];
        #pragma unroll
        for (int r = 0; r < 8; ++r) {
            float v = fmaxf(fmaxf(st[0][r], st[1][r]), fmaxf(st[2][r], st[3][r]));
            #pragma unroll
            for (int msk = 1; msk < 16; msk <<= 1)
                v = fmaxf(v, __shfl_xor(v, msk, 32));
            float mn = fmaxf(mrow[r], v);
            al[r] = __expf(mrow[r] - mn);
            mrow[r] = mn;
        }
        float ps[8] = {};
        #pragma unroll
        for (int kt = 0; kt < 4; ++kt)
            #pragma unroll
            for (int r = 0; r < 8; ++r) {
                float p = __expf(st[kt][r] - mrow[r]);
                st[kt][r] = p;
                ps[r] += p;
            }
        #pragma unroll
        for (int r = 0; r < 8; ++r) {
            float v = ps[r];
            #pragma unroll
            for (int msk = 1; msk < 16; msk <<= 1)
                v += __shfl_xor(v, msk, 32);
            lrow[r] = lrow[r] * al[r] + v;
            #pragma unroll
            for (int dt = 0; dt < 4; ++dt) o[dt][r] *= al[r];
        }

        // C-layout -> A-layout for P via per-wave LDS staging (bf16).
        #pragma unroll
        for (int kt = 0; kt < 4; ++kt)
            #pragma unroll
            for (int r = 0; r < 8; ++r) {
                int row = r + 8 * (lane >> 4);
                int col = kt * 16 + (lane & 15);
                myP[row * 64 + col] = f32_to_bf16(st[kt][r]);
            }
        // (DS ops from the same wave are in-order; loads below see the stores.)

        // O += P V
        #pragma unroll
        for (int kc = 0; kc < 2; ++kc) {
            v16bf ap = load_a_frag(myP + (lane & 15) * 64, kc * 32, lane);
            #pragma unroll
            for (int dt = 0; dt < 4; ++dt) {
                const unsigned short* vrow = cV + (dt * 16 + (lane & 15)) * 64;
                v16bf bv = load_b_frag(vrow, kc * 32, lane);
                o[dt] = wmma_bf16(ap, bv, o[dt]);
            }
        }
        __syncthreads();          // all waves done with buffer `cur`
    }

    // Epilogue: out[b][s][h*64+d] = O / l
    #pragma unroll
    for (int dt = 0; dt < 4; ++dt)
        #pragma unroll
        for (int r = 0; r < 8; ++r) {
            int s = qblk * 128 + wave * 16 + r + 8 * (lane >> 4);
            int d = dt * 16 + (lane & 15);
            out[((size_t)b * SSZ + s) * NE + h * DH + d] = o[dt][r] / lrow[r];
        }
}

// ---------------- Host launcher ----------------
extern "C" void kernel_launch(void* const* d_in, const int* in_sizes, int n_in,
                              void* d_out, int out_size, void* d_ws, size_t ws_size,
                              hipStream_t stream) {
    const float* x = (const float*)d_in[0];       // [2,2048,1024] f32
    const float* w = (const float*)d_in[1];       // [3,1024,1024] f32
    float* out = (float*)d_out;                   // [2,2048,1024] f32

    char* ws = (char*)d_ws;
    const size_t xb_bytes = (size_t)BSZ * SSZ * DSZ * 2;      // 8 MiB
    const size_t wT_bytes = (size_t)3 * NE * DSZ * 2;         // 6 MiB
    const size_t qk_bytes = (size_t)BSZ * HN * SSZ * DH * 2;  // 8 MiB each
    unsigned short* xb = (unsigned short*)ws;
    unsigned short* wT = (unsigned short*)(ws + xb_bytes);
    unsigned short* Qb = (unsigned short*)(ws + xb_bytes + wT_bytes);
    unsigned short* Kb = (unsigned short*)(ws + xb_bytes + wT_bytes + qk_bytes);
    unsigned short* Vt = (unsigned short*)(ws + xb_bytes + wT_bytes + 2 * qk_bytes);

    const int nx = BSZ * SSZ * DSZ;               // 4,194,304
    cvt_x_kernel<<<(nx + 255) / 256, 256, 0, stream>>>(x, xb, nx);
    cvt_w_kernel<<<(3 * NE * DSZ) / 256, 256, 0, stream>>>(w, wT);

    proj_kernel<<<dim3((BSZ * SSZ) / 128, NE / 64, 3), 256, 0, stream>>>(
        xb, wT, Qb, Kb, Vt);

    attn_kernel<<<dim3(SSZ / 128, HN, BSZ), 256, 0, stream>>>(Qb, Kb, Vt, out);
}
